// LinearCrossAttention_89446988906572
// MI455X (gfx1250) — matrix operand order repaired and checked by reading, not compiled, not run on hardware
//
#include <hip/hip_runtime.h>
#include <hip/hip_bf16.h>
#include <stdint.h>

typedef _Float16 h8   __attribute__((ext_vector_type(8)));
typedef _Float16 h4   __attribute__((ext_vector_type(4)));
typedef _Float16 v16h __attribute__((ext_vector_type(16)));
typedef float    v8f  __attribute__((ext_vector_type(8)));

// LDS tile geometry: 128 rows x 32 k, padded to 40 f16 (80B) per row so that
// consecutive rows land on different bank groups (80B/4 = 20 banks stride).
#define TILE_ROWS 128
#define TILE_K    32
#define ROW_PAD   40   // f16 elements per padded row

// ---------------------------------------------------------------------------
// A-fragment: 16-bit A matrix 16x32 (MxK), ISA interleaved layout:
// lane<16: K = kbase+0..7 and kbase+16..23 (kbase=0); lane>=16: kbase=8.
__device__ __forceinline__ v16h load_a_frag(const _Float16* p) {
  h8 lo = *(const h8*)p;
  h8 hi = *(const h8*)(p + 16);
  return __builtin_shufflevector(lo, hi,
      0,1,2,3,4,5,6,7,8,9,10,11,12,13,14,15);
}
// B-fragment: 16-bit B matrix 32x16 (KxN), interleaved layout (same as A):
// lane<16: K = 0..7 and 16..23; lane>=16: K = 8..15 and 24..31.
__device__ __forceinline__ v16h load_b_frag(const _Float16* p) {
  h8 lo = *(const h8*)p;
  h8 hi = *(const h8*)(p + 16);
  return __builtin_shufflevector(lo, hi,
      0,1,2,3,4,5,6,7,8,9,10,11,12,13,14,15);
}

__device__ __forceinline__ unsigned lds_off_u32(const void* p) {
  // Generic pointer to __shared__: low 32 bits are the LDS byte offset.
  return (unsigned)(uintptr_t)p;
}

__device__ __forceinline__ void wait_async0() {
#if __has_builtin(__builtin_amdgcn_s_wait_asynccnt)
  __builtin_amdgcn_s_wait_asynccnt(0);
#else
  asm volatile("s_wait_asynccnt 0x0" ::: "memory");
#endif
}

// Async-copy one 128x32 f16 tile (global row stride ldg elements) into LDS
// (padded row stride ROW_PAD). 256 threads, 2x 16B chunks each (512 chunks).
__device__ __forceinline__ void tile_to_lds_async(unsigned ldsBase,
                                                  const _Float16* g,
                                                  long long ldg) {
  const int t = threadIdx.x;
  #pragma unroll
  for (int j = 0; j < 2; ++j) {
    const int c   = t * 2 + j;   // 0..511
    const int row = c >> 2;      // 0..127
    const int seg = c & 3;       // 16B segment within 64B row
    const unsigned lds_off = ldsBase + (unsigned)(row * (ROW_PAD * 2) + seg * 16);
    const unsigned long long ga =
        (unsigned long long)(uintptr_t)(g + (long long)row * ldg + seg * 8);
    asm volatile("global_load_async_to_lds_b128 %0, %1, off"
                 :: "v"(lds_off), "v"(ga) : "memory");
  }
}

// ---------------------------------------------------------------------------
// Generic WMMA GEMM:  C[M,N] = A[M,K] * Bt[N,K]^T (+ bias[n])
//   mode 0: C f32 row-major (ldc), optional bias
//   mode 1: C f16 stored transposed: C[n*ldc + m]
// Batched via gridDim.y: bh -> b=bh>>3, h=bh&7, with element strides.
// Block: 256 threads (8 waves). Block tile 128x128, wave tile 32x64 (2x4).
// Double-buffered async-to-LDS staging of 128x32 A/B tiles.
__global__ __launch_bounds__(256)
void gemm_wmma_kernel(const _Float16* __restrict__ A, long long lda,
                      const _Float16* __restrict__ Bt, long long ldb,
                      void* __restrict__ Cv, long long ldc,
                      int M, int N, int K,
                      const float* __restrict__ bias, int mode,
                      long long sAb, long long sAh,
                      long long sBb, long long sBh,
                      long long sCb, long long sCh)
{
  (void)M;
  __shared__ _Float16 lA[2][TILE_ROWS * ROW_PAD];
  __shared__ _Float16 lB[2][TILE_ROWS * ROW_PAD];

  const int lane   = threadIdx.x & 31;
  const int wave   = threadIdx.x >> 5;
  const int wm     = wave >> 1;     // 0..3  (M direction)
  const int wn     = wave & 1;      // 0..1  (N direction)
  const int rl     = lane & 15;
  const int hiHalf = lane >> 4;     // 0 or 1
  const int nTilesN = N >> 7;
  const int bm = (blockIdx.x / nTilesN) << 7;
  const int bn = (blockIdx.x % nTilesN) << 7;
  const int bh = blockIdx.y;
  const int b = bh >> 3, h = bh & 7;
  const _Float16* Ab = A  + (long long)b * sAb + (long long)h * sAh + (long long)bm * lda;
  const _Float16* Bb = Bt + (long long)b * sBb + (long long)h * sBh + (long long)bn * ldb;

  const unsigned aBase0 = lds_off_u32(&lA[0][0]);
  const unsigned aBase1 = lds_off_u32(&lA[1][0]);
  const unsigned bBase0 = lds_off_u32(&lB[0][0]);
  const unsigned bBase1 = lds_off_u32(&lB[1][0]);

  v8f acc[2][4];
  #pragma unroll
  for (int i = 0; i < 2; ++i)
    #pragma unroll
    for (int j = 0; j < 4; ++j)
      acc[i][j] = v8f{0.f,0.f,0.f,0.f,0.f,0.f,0.f,0.f};

  const int kA = hiHalf * 8;    // A k sub-offset per lane half (interleaved)
  const int kB = hiHalf * 8;    // B k sub-offset per lane half (interleaved)

  // Prologue: stage first K-slice into buffer 0.
  tile_to_lds_async(aBase0, Ab, lda);
  tile_to_lds_async(bBase0, Bb, ldb);

  int buf = 0;
  for (int k0 = 0; k0 < K; k0 += TILE_K, buf ^= 1) {
    wait_async0();        // our async writes into current buffer are done
    __syncthreads();      // everyone's writes are done; prev compute finished
    if (k0 + TILE_K < K) {
      const unsigned an = buf ? aBase0 : aBase1;
      const unsigned bn2 = buf ? bBase0 : bBase1;
      tile_to_lds_async(an,  Ab + (k0 + TILE_K), lda);
      tile_to_lds_async(bn2, Bb + (k0 + TILE_K), ldb);
    }
    const _Float16* sA = buf ? &lA[1][0] : &lA[0][0];
    const _Float16* sB = buf ? &lB[1][0] : &lB[0][0];

    v16h af[2], bf[4];
    #pragma unroll
    for (int mf = 0; mf < 2; ++mf) {
      const int r = wm * 32 + mf * 16 + rl;
      af[mf] = load_a_frag(sA + r * ROW_PAD + kA);
    }
    #pragma unroll
    for (int nf = 0; nf < 4; ++nf) {
      const int c = wn * 64 + nf * 16 + rl;
      bf[nf] = load_b_frag(sB + c * ROW_PAD + kB);
    }
    #pragma unroll
    for (int mf = 0; mf < 2; ++mf)
      #pragma unroll
      for (int nf = 0; nf < 4; ++nf)
        acc[mf][nf] = __builtin_amdgcn_wmma_f32_16x16x32_f16(
            false, af[mf], false, bf[nf], (short)0, acc[mf][nf], false, false);
  }

  // C/D layout: VGPR r: lanes 0-15 -> M=r, lanes 16-31 -> M=8+r; N=lane&15
  const int mBase = bm + wm * 32 + hiHalf * 8;
  if (mode == 0) {
    float* C = (float*)Cv + (long long)b * sCb + (long long)h * sCh;
    #pragma unroll
    for (int mf = 0; mf < 2; ++mf) {
      #pragma unroll
      for (int nf = 0; nf < 4; ++nf) {
        const int n  = bn + wn * 64 + nf * 16 + rl;
        const float bv = bias ? bias[n] : 0.0f;
        const int m0 = mBase + mf * 16;
        #pragma unroll
        for (int r = 0; r < 8; ++r)
          C[(long long)(m0 + r) * ldc + n] = acc[mf][nf][r] + bv;
      }
    }
  } else {
    _Float16* C = (_Float16*)Cv + (long long)b * sCb + (long long)h * sCh;
    #pragma unroll
    for (int mf = 0; mf < 2; ++mf) {
      #pragma unroll
      for (int nf = 0; nf < 4; ++nf) {
        const int n  = bn + wn * 64 + nf * 16 + rl;
        const int m0 = mBase + mf * 16;
        #pragma unroll
        for (int r = 0; r < 8; ++r)
          C[(long long)n * ldc + (m0 + r)] = (_Float16)acc[mf][nf][r];
      }
    }
  }
}

// ---------------------------------------------------------------------------
// Weight convert + transpose: W[K,N] f32 -> WT[N,K] f16
__global__ __launch_bounds__(256)
void wtrans_kernel(const float* __restrict__ W, _Float16* __restrict__ WT,
                   int Kd, int Nd)
{
  long long idx = (long long)blockIdx.x * 256 + threadIdx.x;
  if (idx >= (long long)Kd * Nd) return;
  int i = (int)(idx / Nd);
  int o = (int)(idx % Nd);
  WT[(long long)o * Kd + i] = (_Float16)W[idx];
}

// ---------------------------------------------------------------------------
// Row LayerNorm: out[row, :Dd] = LN(x[row, :Dd]) in f16. One block per row.
__global__ __launch_bounds__(256)
void ln_kernel(const float* __restrict__ x, const float* __restrict__ g,
               const float* __restrict__ bb, _Float16* __restrict__ out, int Dd)
{
  __shared__ float s1[256];
  __shared__ float s2[256];
  const int tid = threadIdx.x;
  const long long base = (long long)blockIdx.x * Dd;
  const float* xr = x + base;
  float a = 0.f, q = 0.f;
  for (int i = tid; i < Dd; i += 256) { float v = xr[i]; a += v; q += v * v; }
  s1[tid] = a; s2[tid] = q;
  __syncthreads();
  for (int off = 128; off > 0; off >>= 1) {
    if (tid < off) { s1[tid] += s1[tid + off]; s2[tid] += s2[tid + off]; }
    __syncthreads();
  }
  const float mean = s1[0] / (float)Dd;
  const float var  = s2[0] / (float)Dd - mean * mean;
  const float rstd = rsqrtf(var + 1e-5f);
  _Float16* orow = out + base;
  for (int i = tid; i < Dd; i += 256)
    orow[i] = (_Float16)((xr[i] - mean) * rstd * g[i] + bb[i]);
}

// ---------------------------------------------------------------------------
// Softmax over last-128 groups (head_dim). One wave per group of 128 f32.
__global__ __launch_bounds__(256)
void qsoftmax_kernel(const float* __restrict__ q, _Float16* __restrict__ out,
                     long long nGroups)
{
  long long gid = ((long long)blockIdx.x * blockDim.x + threadIdx.x) >> 5;
  if (gid >= nGroups) return;
  const int lane = threadIdx.x & 31;
  const float4 v = *(const float4*)(q + gid * 128 + lane * 4);
  float m = fmaxf(fmaxf(v.x, v.y), fmaxf(v.z, v.w));
  #pragma unroll
  for (int off = 16; off > 0; off >>= 1) m = fmaxf(m, __shfl_xor(m, off, 32));
  const float e0 = __expf(v.x - m), e1 = __expf(v.y - m);
  const float e2 = __expf(v.z - m), e3 = __expf(v.w - m);
  float s = e0 + e1 + e2 + e3;
  #pragma unroll
  for (int off = 16; off > 0; off >>= 1) s += __shfl_xor(s, off, 32);
  const float inv = 1.0f / s;
  h4 r;
  r[0] = (_Float16)(e0 * inv); r[1] = (_Float16)(e1 * inv);
  r[2] = (_Float16)(e2 * inv); r[3] = (_Float16)(e3 * inv);
  *(h4*)(out + gid * 128 + lane * 4) = r;
}

// ---------------------------------------------------------------------------
// k softmax over N axis (axis=1), one thread per (b,d) column; writes
// transposed f16: kT[(b*8+h)*128 + kk][n]  (row length Nn).
__global__ __launch_bounds__(256)
void ksoftmax_kernel(const float* __restrict__ k, _Float16* __restrict__ kT,
                     int Bc, int Nn, int Dd)
{
  const int idx = blockIdx.x * 256 + threadIdx.x;
  if (idx >= Bc * Dd) return;
  const int b = idx / Dd, d = idx % Dd;
  const float* col = k + (long long)b * Nn * Dd + d;
  float m = -3.402823466e38f;
  for (int n = 0; n < Nn; ++n) m = fmaxf(m, col[(long long)n * Dd]);
  float s = 0.f;
  for (int n = 0; n < Nn; ++n) s += __expf(col[(long long)n * Dd] - m);
  const float inv = 1.0f / s;
  const int h = d >> 7, kk = d & 127;
  _Float16* outp = kT + ((long long)(b * 8 + h) * 128 + kk) * Nn;
  for (int n = 0; n < Nn; ++n)
    outp[n] = (_Float16)(__expf(col[(long long)n * Dd] - m) * inv);
}

// ---------------------------------------------------------------------------
// v f32 [b,n,d] -> vT f16 [(b*8+h)*128+vv][n]  (row length 512)
__global__ __launch_bounds__(256)
void vtrans_kernel(const float* __restrict__ v, _Float16* __restrict__ vT)
{
  const long long idx = (long long)blockIdx.x * 256 + threadIdx.x; // < 4*512*1024
  const int d = (int)(idx & 1023);
  const long long bn = idx >> 10;
  const int n = (int)(bn & 511);
  const int b = (int)(bn >> 9);
  const int h = d >> 7, vv = d & 127;
  vT[(((long long)(b * 8 + h) * 128 + vv) << 9) + n] = (_Float16)v[idx];
}

// ---------------------------------------------------------------------------
extern "C" void kernel_launch(void* const* d_in, const int* in_sizes, int n_in,
                              void* d_out, int out_size, void* d_ws, size_t ws_size,
                              hipStream_t stream) {
  (void)in_sizes; (void)n_in; (void)out_size; (void)ws_size;
  const float* x     = (const float*)d_in[0];   // [4,4096,1024]
  const float* cond  = (const float*)d_in[1];   // [4,512,768]
  const float* ln_g  = (const float*)d_in[2];
  const float* ln_b  = (const float*)d_in[3];
  const float* tln_g = (const float*)d_in[4];
  const float* tln_b = (const float*)d_in[5];
  const float* Wq    = (const float*)d_in[6];   // [1024,1024]
  const float* bq    = (const float*)d_in[7];
  const float* Wk    = (const float*)d_in[8];   // [768,1024]
  const float* bk    = (const float*)d_in[9];
  const float* Wv    = (const float*)d_in[10];  // [768,1024]
  const float* bv    = (const float*)d_in[11];
  float* y = (float*)d_out;

  constexpr int B = 4, T = 4096, NC = 512, D = 1024, L = 768;
  constexpr long long BT = (long long)B * T;   // 16384
  constexpr long long BN = (long long)B * NC;  // 2048

  char* ws = (char*)d_ws;
  size_t off = 0;
  auto alloc = [&](size_t bytes) -> void* {
    void* p = ws + off;
    off += (bytes + 255) & ~(size_t)255;
    return p;
  };
  _Float16* WqT = (_Float16*)alloc((size_t)D * D * 2);
  _Float16* WkT = (_Float16*)alloc((size_t)L * D * 2);
  _Float16* WvT = (_Float16*)alloc((size_t)L * D * 2);
  _Float16* xn  = (_Float16*)alloc((size_t)BT * D * 2);
  _Float16* cn  = (_Float16*)alloc((size_t)BN * L * 2);
  float*    qf  = (float*)   alloc((size_t)BT * D * 4);
  _Float16* qs  = (_Float16*)alloc((size_t)BT * D * 2);
  float*    kf  = (float*)   alloc((size_t)BN * D * 4);
  float*    vf  = (float*)   alloc((size_t)BN * D * 4);
  _Float16* kT  = (_Float16*)alloc((size_t)32 * 128 * 512 * 2);
  _Float16* vT  = (_Float16*)alloc((size_t)32 * 128 * 512 * 2);
  _Float16* ctxT= (_Float16*)alloc((size_t)32 * 128 * 128 * 2);

  // 1) weight convert+transpose
  wtrans_kernel<<<dim3((D * D + 255) / 256), 256, 0, stream>>>(Wq, WqT, D, D);
  wtrans_kernel<<<dim3((L * D + 255) / 256), 256, 0, stream>>>(Wk, WkT, L, D);
  wtrans_kernel<<<dim3((L * D + 255) / 256), 256, 0, stream>>>(Wv, WvT, L, D);

  // 2) layernorms -> f16
  ln_kernel<<<dim3((unsigned)BT), 256, 0, stream>>>(x, ln_g, ln_b, xn, D);
  ln_kernel<<<dim3((unsigned)BN), 256, 0, stream>>>(cond, tln_g, tln_b, cn, L);

  // 3) projections
  // q = xn @ Wq + bq  : M=16384 N=1024 K=1024
  gemm_wmma_kernel<<<dim3((BT >> 7) * (D >> 7), 1), 256, 0, stream>>>(
      xn, D, WqT, D, qf, D, (int)BT, D, D, bq, 0, 0,0, 0,0, 0,0);
  // k = cn @ Wk + bk  : M=2048 N=1024 K=768
  gemm_wmma_kernel<<<dim3((BN >> 7) * (D >> 7), 1), 256, 0, stream>>>(
      cn, L, WkT, L, kf, D, (int)BN, D, L, bk, 0, 0,0, 0,0, 0,0);
  // v = cn @ Wv + bv
  gemm_wmma_kernel<<<dim3((BN >> 7) * (D >> 7), 1), 256, 0, stream>>>(
      cn, L, WvT, L, vf, D, (int)BN, D, L, bv, 0, 0,0, 0,0, 0,0);

  // 4) softmaxes + transposes
  qsoftmax_kernel<<<dim3((unsigned)(BT * 8 / 8)), 256, 0, stream>>>(qf, qs, BT * 8);
  ksoftmax_kernel<<<dim3((B * D + 255) / 256), 256, 0, stream>>>(kf, kT, B, NC, D);
  vtrans_kernel<<<dim3((unsigned)(BN * D / 256)), 256, 0, stream>>>(vf, vT);

  // 5) context[b,h] = kT(128x512) * vT^T -> stored transposed f16 [vv][kk]
  //    M=128 N=128 K=512, batched over 32 (b,h)
  gemm_wmma_kernel<<<dim3(1, 32), 256, 0, stream>>>(
      kT, 512, vT, 512, ctxT, 128, 128, 128, 512, nullptr, 1,
      /*sAb*/ (long long)8 * 128 * 512, /*sAh*/ (long long)128 * 512,
      /*sBb*/ (long long)8 * 128 * 512, /*sBh*/ (long long)128 * 512,
      /*sCb*/ (long long)8 * 128 * 128, /*sCh*/ (long long)128 * 128);

  // 6) y[b,:,h*128+vv] = q_s[b,:,h*128+:] @ context  : M=4096 N=128 K=128
  gemm_wmma_kernel<<<dim3(T >> 7, 32), 256, 0, stream>>>(
      qs, D, ctxT, 128, y, D, T, 128, 128, nullptr, 0,
      /*sAb*/ (long long)T * D, /*sAh*/ 128,
      /*sBb*/ (long long)8 * 128 * 128, /*sBh*/ (long long)128 * 128,
      /*sCb*/ (long long)T * D, /*sCh*/ 128);
}
